// NeuralAstarField_31722628448273
// MI455X (gfx1250) — matrix operand order repaired and checked by reading, not compiled, not run on hardware
//
#include <hip/hip_runtime.h>

typedef __attribute__((ext_vector_type(16))) _Float16 v16h;
typedef __attribute__((ext_vector_type(8)))  _Float16 v8h;
typedef __attribute__((ext_vector_type(8)))  float    v8f;
typedef __attribute__((ext_vector_type(4)))  unsigned int u32x4;
typedef __attribute__((ext_vector_type(8)))  int i32x8;
typedef __attribute__((ext_vector_type(4)))  int i32x4;

#define BB   16
#define NPIX 4096            // 64*64
#define NTOT (BB * NPIX)     // 65536
#define TSTEPS 512

#if __has_builtin(__builtin_amdgcn_tensor_load_to_lds)
#define HAVE_TDM 1
#else
#define HAVE_TDM 0
#endif

static __device__ __forceinline__ float bn_scale(float gmv) {
    return gmv * rsqrtf(1.0f + 1e-5f);
}

#if HAVE_TDM
// Tensor Data Mover: stage a 2-D tile (nrows x row_dwords, 4-byte units,
// row-stride == row_dwords) from global memory into LDS at lds_off.
// D# layout per CDNA5 ISA; groups 2/3/4
// zero (<=2D tensor). Completion tracked with TENSORcnt.
// This toolchain exposes the 6-arg builtin form:
//   (u32x4 g0, i32x8 g1, i32x4, i32x4, i32x8, i32 cpol)
static __device__ __forceinline__ void tdm_load_2d_to_lds(
        unsigned int lds_off, unsigned long long gaddr,
        unsigned int row_dwords, unsigned int nrows) {
    u32x4 g0;
    g0[0] = 1u;                                        // count=1, user desc
    g0[1] = lds_off;                                   // lds_addr (bytes)
    g0[2] = (unsigned int)gaddr;                       // global_addr[31:0]
    g0[3] = (unsigned int)((gaddr >> 32) & 0x01FFFFFFu) | (2u << 30); // [56:32], type=2
    i32x8 g1;
    g1[0] = (int)(2u << 16);                           // data_size=2 (4B), mask=0
    g1[1] = (int)((row_dwords & 0xFFFFu) << 16);       // tensor_dim0[15:0] -> bits 63:48
    g1[2] = (int)(((row_dwords >> 16) & 0xFFFFu)       // tensor_dim0[31:16]
                | ((nrows & 0xFFFFu) << 16));          // tensor_dim1[15:0]
    g1[3] = (int)(((nrows >> 16) & 0xFFFFu)            // tensor_dim1[31:16]
                | ((row_dwords & 0xFFFFu) << 16));     // tile_dim0
    g1[4] = (int)(nrows & 0xFFFFu);                    // tile_dim1 (tile_dim2=0)
    g1[5] = (int)row_dwords;                           // tensor_dim0_stride[31:0]
    g1[6] = 0;                                         // stride hi, dim1_stride lo
    g1[7] = 0;
    i32x4 z4 = {0, 0, 0, 0};
    i32x8 z8 = {0, 0, 0, 0, 0, 0, 0, 0};
    __builtin_amdgcn_tensor_load_to_lds(g0, g1, z4, z4, z8, 0);
}
#endif

// ---------------------------------------------------------------------------
// Weight transform: w[cout][cin][3][3] (f32) -> wt[cout][p*Cin + c] (f16)
// ---------------------------------------------------------------------------
__global__ void wt_transform_kernel(const float* __restrict__ w,
                                    _Float16* __restrict__ wt,
                                    int Cin, int Cout) {
    int K = 9 * Cin;
    int idx = blockIdx.x * blockDim.x + threadIdx.x;
    if (idx >= Cout * K) return;
    int cout = idx / K;
    int r = idx - cout * K;
    int p = r / Cin;
    int c = r - p * Cin;
    wt[idx] = (_Float16)w[cout * Cin * 9 + c * 9 + p];
}

// ---------------------------------------------------------------------------
// Layer 0: 3 -> 32, scalar (K=27 too small for WMMA). BN+ReLU fused.
// Output act1 in NHWC f16.
// ---------------------------------------------------------------------------
__global__ void conv0_kernel(const float* __restrict__ xm,
                             const float* __restrict__ xs,
                             const float* __restrict__ xg,
                             const float* __restrict__ w,
                             const float* __restrict__ bc,
                             const float* __restrict__ gm,
                             const float* __restrict__ bt,
                             _Float16* __restrict__ out) {
    int i = blockIdx.x * blockDim.x + threadIdx.x;
    if (i >= NTOT) return;
    int b = i >> 12, y = (i >> 6) & 63, x = i & 63;
    const float* chans[3] = { xm + b * NPIX, xs + b * NPIX, xg + b * NPIX };
    float patch[27];
    #pragma unroll
    for (int c = 0; c < 3; c++)
        #pragma unroll
        for (int dy = 0; dy < 3; dy++)
            #pragma unroll
            for (int dx = 0; dx < 3; dx++) {
                int yy = y + dy - 1, xx = x + dx - 1;
                bool ok = ((unsigned)yy < 64u) && ((unsigned)xx < 64u);
                patch[c * 9 + dy * 3 + dx] = ok ? chans[c][yy * 64 + xx] : 0.0f;
            }
    for (int o = 0; o < 32; o++) {
        float acc = bc[o];
        #pragma unroll
        for (int k = 0; k < 27; k++) acc += patch[k] * w[o * 27 + k];
        float v = acc * bn_scale(gm[o]) + bt[o];
        out[(size_t)i * 32 + o] = (_Float16)fmaxf(v, 0.0f);
    }
}

// ---------------------------------------------------------------------------
// Layers 1-3: implicit-GEMM conv via V_WMMA_F32_16X16X32_F16.
// One wave per 16-pixel x 16-cout tile. Activations NHWC f16. The 16 weight
// rows [16][9*Cin] f16 of this ctile are DMA'd into LDS by the Tensor Data
// Mover, then B fragments come from LDS (ds_load). BN+ReLU fused in epilogue.
// Templated Cin/Cout -> fully unrolled K loop (9*Cin/32 WMMAs back-to-back).
// ---------------------------------------------------------------------------
template <int Cin, int Cout>
__global__ __launch_bounds__(32) void conv_wmma_kernel(
        const _Float16* __restrict__ actin, const _Float16* __restrict__ wt,
        const float* __restrict__ bc, const float* __restrict__ gm,
        const float* __restrict__ bt, _Float16* __restrict__ actout) {
    constexpr int K = 9 * Cin;
    int mt = blockIdx.x;                       // 0..4095 pixel tiles
    int b = mt >> 8;
    int rem = mt & 255;
    int y = rem >> 2;
    int x0 = (rem & 3) << 4;
    int ctile = blockIdx.y;
    int tid = threadIdx.x;
    int l = tid & 15;
    int hi = tid >> 4;

#if HAVE_TDM
    __shared__ __align__(32) _Float16 lds_wt[16 * K];
    // Wave-level DMA: 16 rows x K halves (= K/2 dwords per row), contiguous.
    tdm_load_2d_to_lds((unsigned int)(size_t)&lds_wt[0],
                       (unsigned long long)(size_t)(wt + (size_t)ctile * 16 * K),
                       (unsigned int)(K / 2), 16u);
#endif

    v8f acc = {};
#if HAVE_TDM
    __builtin_amdgcn_s_wait_tensorcnt(0);      // TDM done before LDS reads
    const _Float16* wrow = &lds_wt[(size_t)l * K];
#else
    const _Float16* wrow = wt + (size_t)(ctile * 16 + l) * K;
#endif

    #pragma unroll
    for (int p = 0; p < 9; p++) {
        int dy = p / 3 - 1, dx = p % 3 - 1;
        int yy = y + dy, xx = x0 + l + dx;
        bool ok = ((unsigned)yy < 64u) && ((unsigned)xx < 64u);
        int rowoff = ok ? ((b * 64 + yy) * 64 + xx) * Cin : 0;  // keep ptr valid
        const _Float16* abase = actin + rowoff;
        #pragma unroll
        for (int c0 = 0; c0 < Cin; c0 += 32) {
            v8h z = {};
            // A frag (16x32, contiguous k, same order as B): lane holds row M=l,
            // K = hi*16 + 0..15 in elems 0..15
            v8h alo = ok ? *(const v8h*)(abase + c0 + hi * 16) : z;
            v8h ahi = ok ? *(const v8h*)(abase + c0 + hi * 16 + 8) : z;
            v16h af = __builtin_shufflevector(alo, ahi,
                          0, 1, 2, 3, 4, 5, 6, 7, 8, 9, 10, 11, 12, 13, 14, 15);
            // B frag (32x16): lane holds column N=l, K = hi*16 + 0..15
            v16h bf = *(const v16h*)(wrow + p * Cin + c0 + hi * 16);
            acc = __builtin_amdgcn_wmma_f32_16x16x32_f16(
                      false, af, false, bf, (short)0, acc, false, false);
        }
    }

    int cout = ctile * 16 + l;
    float sc = bn_scale(gm[cout]);
    float bias = bc[cout] * sc + bt[cout];
    #pragma unroll
    for (int r = 0; r < 8; r++) {                 // C/D: VGPR r -> M = r + hi*8
        int x = x0 + r + hi * 8;
        float v = acc[r] * sc + bias;
        v = fmaxf(v, 0.0f);
        actout[(size_t)((b * 64 + y) * 64 + x) * Cout + cout] = (_Float16)v;
    }
}

// ---------------------------------------------------------------------------
// Layer 4 (256 -> 1, no ReLU) + the three 1x1 heads, fused.
// ---------------------------------------------------------------------------
__global__ void conv4_head_kernel(const _Float16* __restrict__ act,
                                  const _Float16* __restrict__ wt,
                                  const float* __restrict__ b4,
                                  const float* __restrict__ gm4,
                                  const float* __restrict__ bt4,
                                  const float* __restrict__ cw, const float* __restrict__ cb,
                                  const float* __restrict__ gw, const float* __restrict__ gb,
                                  const float* __restrict__ ow, const float* __restrict__ ob,
                                  float* __restrict__ cost,
                                  float* __restrict__ geo,
                                  float* __restrict__ obs) {
    int i = blockIdx.x * blockDim.x + threadIdx.x;
    if (i >= NTOT) return;
    int b = i >> 12, y = (i >> 6) & 63, x = i & 63;
    float acc = 0.0f;
    for (int p = 0; p < 9; p++) {
        int dy = p / 3 - 1, dx = p % 3 - 1;
        int yy = y + dy, xx = x + dx;
        if ((unsigned)yy >= 64u || (unsigned)xx >= 64u) continue;
        const _Float16* base = act + (size_t)((b * 64 + yy) * 64 + xx) * 256;
        const _Float16* wbase = wt + p * 256;
        #pragma unroll 4
        for (int c = 0; c < 256; c += 8) {
            v8h av = *(const v8h*)(base + c);
            v8h wv = *(const v8h*)(wbase + c);
            #pragma unroll
            for (int e = 0; e < 8; e++) acc += (float)av[e] * (float)wv[e];
        }
    }
    float feat = (acc + b4[0]) * bn_scale(gm4[0]) + bt4[0];
    cost[i] = 1.0f / (1.0f + expf(-(feat * cw[0] + cb[0])));
    geo[i]  = fmaxf(feat * gw[0] + gb[0], 0.0f);
    obs[i]  = fmaxf(feat * ow[0] + ob[0], 0.0f);
}

// ---------------------------------------------------------------------------
// A* setup: locate goal one-hot, build h = chebyshev + 1e-3*euclid + cost,
// init g/open/hist/parents and the incremental argmax key.
// ---------------------------------------------------------------------------
__global__ void find_goal_kernel(const float* __restrict__ goal, int* __restrict__ gidx) {
    int b = blockIdx.x;
    for (int k = threadIdx.x; k < NPIX; k += blockDim.x)
        if (goal[b * NPIX + k] > 0.5f) gidx[b] = k;   // exactly one writer per b
}

__global__ void astar_init_kernel(const float* __restrict__ start,
                                  const float* __restrict__ cost,
                                  const int* __restrict__ gidx,
                                  float* __restrict__ h, float* __restrict__ g,
                                  float* __restrict__ open, float* __restrict__ key,
                                  int* __restrict__ parents, float* __restrict__ hist) {
    int i = blockIdx.x * blockDim.x + threadIdx.x;
    if (i >= NTOT) return;
    int b = i >> 12, r = (i >> 6) & 63, c = i & 63;
    int gi = gidx[b], gr = gi >> 6, gc = gi & 63;
    float dr = fabsf((float)r - (float)gr), dc = fabsf((float)c - (float)gc);
    float cheb = fmaxf(dr, dc);                       // sum - min of 2 = max
    float euc = sqrtf(dr * dr + dc * dc);
    float hv = cheb + 0.001f * euc + cost[i];
    h[i] = hv;
    g[i] = 0.0f;
    float op = start[i];
    open[i] = op;
    key[i] = expf(-0.5f * hv * 0.125f) * op;          // exp(-(.5g+.5h)/8), g=0
    parents[i] = gi;
    hist[i] = 0.0f;
}

// ---------------------------------------------------------------------------
// 512 sequential A* steps in one workgroup (32 wave32 waves; 64 thr/batch).
// Forward st_softmax is an exact hard argmax; sel/hist/open/idx are binary,
// so each step = per-batch argmax of key + O(8) cell updates.
// ---------------------------------------------------------------------------
__global__ __launch_bounds__(1024) void astar_kernel(
        const float* __restrict__ goal, const float* __restrict__ obstacles,
        const float* __restrict__ cost, const float* __restrict__ h,
        float* __restrict__ g, float* __restrict__ open, float* __restrict__ hist,
        int* __restrict__ parents, float* __restrict__ key, int* __restrict__ tlast) {
    __shared__ float redv[1024];
    __shared__ int   redi[1024];
    __shared__ int   s_unsolved[16];
    __shared__ int   s_done;
    __shared__ int   s_tlast;
    int tid = threadIdx.x;
    int b = tid >> 6, lane = tid & 63;
    if (tid == 0) { s_done = 0; s_tlast = 0; }
    __syncthreads();

    for (int t = 0; t < TSTEPS; ++t) {
        // --- argmax(key) per batch, lowest-index tie break (matches jnp) ---
        float bestv = -1.0f; int besti = 0;
        const float* kb = key + b * NPIX;
        for (int k = lane; k < NPIX; k += 64) {
            float v = kb[k];
            if (v > bestv) { bestv = v; besti = k; }
        }
        redv[tid] = bestv; redi[tid] = besti;
        __syncthreads();
        for (int s = 32; s > 0; s >>= 1) {
            if (lane < s) {
                float ov = redv[tid + s]; int oi = redi[tid + s];
                if (ov > redv[tid] || (ov == redv[tid] && oi < redi[tid])) {
                    redv[tid] = ov; redi[tid] = oi;
                }
            }
            __syncthreads();
        }
        // --- O(1) state update by one thread per batch ---
        if (lane == 0) {
            int ind = redi[tid];
            int base = b * NPIX;
            int unsolved = (goal[base + ind] < 1e-8f) ? 1 : 0;
            s_unsolved[b] = unsolved;
            hist[base + ind] = fminf(hist[base + ind] + 1.0f, 1.0f);
            float gi = g[base + ind];
            float oi = open[base + ind];
            if (unsolved) {
                oi = fminf(fmaxf(oi - 1.0f, 0.0f), 1.0f);
                open[base + ind] = oi;
                key[base + ind] = expf(-0.5f * (gi + h[base + ind]) * 0.125f) * oi;
            }
            float gnew = gi + cost[base + ind];
            int iy = ind >> 6, ix = ind & 63;
            for (int dy = -1; dy <= 1; dy++)
                for (int dx = -1; dx <= 1; dx++) {
                    if (dy == 0 && dx == 0) continue;
                    int jy = iy + dy, jx = ix + dx;
                    if ((unsigned)jy >= 64u || (unsigned)jx >= 64u) continue;
                    int j = base + jy * 64 + jx;
                    if (obstacles[j] == 0.0f) continue;           // nb = 0
                    float oj = open[j], hj = hist[j], gj = g[j];
                    float idxv = (1.0f - oj) * (1.0f - hj)
                               + oj * ((gj > gnew) ? 1.0f : 0.0f);
                    if (idxv > 0.5f) {                            // idx == 1
                        g[j] = gnew;
                        open[j] = 1.0f;
                        parents[j] = ind;
                        key[j] = expf(-0.5f * (gnew + h[j]) * 0.125f);
                    }
                }
        }
        __syncthreads();
        if (tid == 0) {
            int alldone = 1;
            for (int bb = 0; bb < 16; bb++) if (s_unsolved[bb]) alldone = 0;
            s_tlast = t;
            if (alldone) s_done = 1;
        }
        __syncthreads();
        if (s_done) break;      // later steps are skipped in the reference
    }
    if (tid == 0) tlast[0] = s_tlast;
}

// ---------------------------------------------------------------------------
// Backtrack: path = goal; walk parents t_last times.
// ---------------------------------------------------------------------------
__global__ __launch_bounds__(1024) void backtrack_kernel(
        const float* __restrict__ goal, const int* __restrict__ parents,
        const int* __restrict__ gidx, const int* __restrict__ tlast,
        float* __restrict__ path) {
    int tid = threadIdx.x;
    for (int i = tid; i < NTOT; i += 1024)
        path[i] = (goal[i] > 0.5f) ? 1.0f : 0.0f;
    __syncthreads();
    if (tid < 16) {
        int b = tid, base = b * NPIX;
        int loc = parents[base + gidx[b]];
        int tl = tlast[0];
        for (int i = 0; i < TSTEPS; i++) {
            if (i < tl) {
                path[base + loc] = 1.0f;
                loc = parents[base + loc];
            }
        }
    }
}

// ---------------------------------------------------------------------------
extern "C" void kernel_launch(void* const* d_in, const int* in_sizes, int n_in,
                              void* d_out, int out_size, void* d_ws, size_t ws_size,
                              hipStream_t stream) {
    (void)in_sizes; (void)n_in; (void)out_size; (void)ws_size;

    const float* xm = (const float*)d_in[0];   // map_designs
    const float* xs = (const float*)d_in[1];   // start_maps
    const float* xg = (const float*)d_in[2];   // goal_maps
    const float* w0 = (const float*)d_in[3],  *b0 = (const float*)d_in[4];
    const float* gm0 = (const float*)d_in[5], *bt0 = (const float*)d_in[6];
    const float* w1 = (const float*)d_in[7],  *b1 = (const float*)d_in[8];
    const float* gm1 = (const float*)d_in[9], *bt1 = (const float*)d_in[10];
    const float* w2 = (const float*)d_in[11], *b2 = (const float*)d_in[12];
    const float* gm2 = (const float*)d_in[13],*bt2 = (const float*)d_in[14];
    const float* w3 = (const float*)d_in[15], *b3 = (const float*)d_in[16];
    const float* gm3 = (const float*)d_in[17],*bt3 = (const float*)d_in[18];
    const float* w4 = (const float*)d_in[19], *b4 = (const float*)d_in[20];
    const float* gm4 = (const float*)d_in[21],*bt4 = (const float*)d_in[22];
    const float* cw = (const float*)d_in[23], *cb = (const float*)d_in[24];
    const float* gw = (const float*)d_in[25], *gb = (const float*)d_in[26];
    const float* ow = (const float*)d_in[27], *ob = (const float*)d_in[28];

    char* ws = (char*)d_ws;
    size_t off = 0;
    auto alloc = [&](size_t bytes) -> void* {
        void* p = ws + off;
        off += (bytes + 255) & ~(size_t)255;
        return p;
    };
    _Float16* act1 = (_Float16*)alloc((size_t)NTOT * 32 * 2);
    _Float16* act2 = (_Float16*)alloc((size_t)NTOT * 64 * 2);
    _Float16* act3 = (_Float16*)alloc((size_t)NTOT * 128 * 2);
    _Float16* act4 = (_Float16*)alloc((size_t)NTOT * 256 * 2);
    _Float16* wt1 = (_Float16*)alloc((size_t)64 * 288 * 2);
    _Float16* wt2 = (_Float16*)alloc((size_t)128 * 576 * 2);
    _Float16* wt3 = (_Float16*)alloc((size_t)256 * 1152 * 2);
    _Float16* wt4 = (_Float16*)alloc((size_t)2304 * 2);
    float* cost  = (float*)alloc((size_t)NTOT * 4);
    float* hbuf  = (float*)alloc((size_t)NTOT * 4);
    float* gbuf  = (float*)alloc((size_t)NTOT * 4);
    float* openb = (float*)alloc((size_t)NTOT * 4);
    float* keyb  = (float*)alloc((size_t)NTOT * 4);
    int*   par   = (int*)alloc((size_t)NTOT * 4);
    int*   gidx  = (int*)alloc(64);
    int*   tlast = (int*)alloc(64);

    float* out  = (float*)d_out;
    float* hist = out;              // output 0
    float* path = out + NTOT;       // output 1
    float* geo  = out + 2 * NTOT;   // output 2
    float* obs  = out + 3 * NTOT;   // output 3

    wt_transform_kernel<<<(64 * 288 + 255) / 256, 256, 0, stream>>>(w1, wt1, 32, 64);
    wt_transform_kernel<<<(128 * 576 + 255) / 256, 256, 0, stream>>>(w2, wt2, 64, 128);
    wt_transform_kernel<<<(256 * 1152 + 255) / 256, 256, 0, stream>>>(w3, wt3, 128, 256);
    wt_transform_kernel<<<(2304 + 255) / 256, 256, 0, stream>>>(w4, wt4, 256, 1);

    conv0_kernel<<<NTOT / 256, 256, 0, stream>>>(xm, xs, xg, w0, b0, gm0, bt0, act1);
    conv_wmma_kernel<32, 64>  <<<dim3(4096, 4),  32, 0, stream>>>(act1, wt1, b1, gm1, bt1, act2);
    conv_wmma_kernel<64, 128> <<<dim3(4096, 8),  32, 0, stream>>>(act2, wt2, b2, gm2, bt2, act3);
    conv_wmma_kernel<128, 256><<<dim3(4096, 16), 32, 0, stream>>>(act3, wt3, b3, gm3, bt3, act4);
    conv4_head_kernel<<<NTOT / 256, 256, 0, stream>>>(act4, wt4, b4, gm4, bt4,
                                                      cw, cb, gw, gb, ow, ob,
                                                      cost, geo, obs);

    find_goal_kernel<<<16, 256, 0, stream>>>(xg, gidx);
    astar_init_kernel<<<NTOT / 256, 256, 0, stream>>>(xs, cost, gidx, hbuf, gbuf,
                                                      openb, keyb, par, hist);
    astar_kernel<<<1, 1024, 0, stream>>>(xg, xm, cost, hbuf, gbuf, openb, hist,
                                         par, keyb, tlast);
    backtrack_kernel<<<1, 1024, 0, stream>>>(xg, par, gidx, tlast, path);
}
